// PermutedDKT_51831665328689
// MI455X (gfx1250) — matrix operand order repaired and checked by reading, not compiled, not run on hardware
//
#include <hip/hip_runtime.h>

// ---------------------------------------------------------------------------
// PermutedDKT on MI455X (gfx1250). ~0.67 TFLOP, GEMM-bound -> bf16 WMMA with
// f32 accumulation. Per-step GRU weights (12MB packed bf16) live in the 192MB
// L2; B tiles are staged to LDS with gfx1250 async copies (ASYNCcnt) and
// shared by all 8 waves of a workgroup. All GEMM k-loops are unrolled by 2
// with ping-pong register sets (no cur<-nxt copies) so loads/async-DMA of
// step k+1 overlap the WMMAs of step k.
// ---------------------------------------------------------------------------

typedef __bf16 bf16;
typedef bf16  v16bf __attribute__((ext_vector_type(16)));
typedef bf16  v8bf  __attribute__((ext_vector_type(8)));
typedef float v8f   __attribute__((ext_vector_type(8)));
typedef __attribute__((address_space(3))) bf16 lds_bf16;

constexpr int kT = 200, kB = 256, kC = 1024, kE = 256;
constexpr int kTB = kT * kB;           // 51200 rows

#define WMMA_BF16(a, b, c) \
  __builtin_amdgcn_wmma_f32_16x16x32_bf16(false, (a), false, (b), (short)0, (c), false, false)

// ---- WMMA fragment loaders (CDNA5 ISA §7.12.2 layouts) ---------------------
// A (16x32 bf16, row-major src): lanes 0-15 row M=lane hold K=[k..k+7],[k+16..k+23];
// lanes 16-31 hold K=[k+8..k+15],[k+24..k+31]  -> two b128 loads per lane.
__device__ __forceinline__ v16bf load_a_frag(const bf16* A, int lda, int mbase,
                                             int kbase, int lane) {
  int row = mbase + (lane & 15);
  int k0  = kbase + ((lane & 16) ? 8 : 0);
  const bf16* p = A + (size_t)row * lda + k0;
  union { v16bf v; v8bf h[2]; } u;
  u.h[0] = *(const v8bf*)(p);
  u.h[1] = *(const v8bf*)(p + 16);
  return u.v;
}
// B (32x16 bf16) pre-packed: a lane's 16 fragment elements are 32 contiguous bytes.
__device__ __forceinline__ v16bf load_b_frag(const bf16* Bp, int ntiles, int kt,
                                             int nt, int lane) {
  const bf16* p = Bp + (((size_t)kt * ntiles + nt) * 32 + lane) * 16;
  union { v16bf v; v8bf h[2]; } u;
  u.h[0] = *(const v8bf*)(p);
  u.h[1] = *(const v8bf*)(p + 8);
  return u.v;
}
// B fragment out of an LDS staging buffer (ds_load_b128 x2).
__device__ __forceinline__ v16bf read_b_lds(const bf16* lbuf, int j, int lane) {
  const bf16* p = lbuf + ((size_t)j * 32 + lane) * 16;
  union { v16bf v; v8bf h[2]; } u;
  u.h[0] = *(const v8bf*)(p);
  u.h[1] = *(const v8bf*)(p + 8);
  return u.v;
}
// 4 B fragments from LDS + 4 WMMAs against one A fragment.
__device__ __forceinline__ void mma4_lds(const bf16* lbuf, v16bf a, int lane, v8f* acc) {
  v16bf b0 = read_b_lds(lbuf, 0, lane);
  v16bf b1 = read_b_lds(lbuf, 1, lane);
  v16bf b2 = read_b_lds(lbuf, 2, lane);
  v16bf b3 = read_b_lds(lbuf, 3, lane);
  acc[0] = WMMA_BF16(a, b0, acc[0]);
  acc[1] = WMMA_BF16(a, b1, acc[1]);
  acc[2] = WMMA_BF16(a, b2, acc[2]);
  acc[3] = WMMA_BF16(a, b3, acc[3]);
}

// ---- gfx1250 async global->LDS copy (16B per thread) -----------------------
__device__ __forceinline__ void stage_async16(const bf16* gsrc, bf16* ldst) {
  unsigned loff = (unsigned)(uintptr_t)(lds_bf16*)ldst;      // LDS address
  unsigned long long ga = (unsigned long long)(uintptr_t)gsrc;
  asm volatile("global_load_async_to_lds_b128 %0, %1, off"
               :: "v"(loff), "v"(ga) : "memory");
}
__device__ __forceinline__ void wait_async0() {
#if __has_builtin(__builtin_amdgcn_s_wait_asynccnt)
  __builtin_amdgcn_s_wait_asynccnt(0);
#else
  asm volatile("s_wait_asynccnt 0x0" ::: "memory");
#endif
}

__device__ __forceinline__ float sigmoidf_(float x) { return 1.f / (1.f + expf(-x)); }

// ---- Sinkhorn + reverse-cumsum (m@L), single 1024-thread workgroup ---------
__global__ void __launch_bounds__(1024) sinkhorn_kernel(
    const float* __restrict__ perm, const int* __restrict__ epoch,
    float* __restrict__ m, bf16* __restrict__ mL) {
  const int r = threadIdx.x;                 // one row / one column per thread
  const int factor = epoch[0] / 10 + 1;
  const float temp = (float)factor * 2.0f;   // INIT_TEMP
  const int unroll = factor * 10;            // INIT_UNROLL
  float mx = -3.402823466e38f;
  for (int c = 0; c < kC; ++c) mx = fmaxf(mx, perm[(size_t)r * kC + c]);
  for (int c = 0; c < kC; ++c)
    m[(size_t)r * kC + c] = expf(temp * (perm[(size_t)r * kC + c] - mx));
  __syncthreads();
  for (int it = 0; it < unroll; ++it) {
    float s = 0.f;
    for (int c = 0; c < kC; ++c) s += m[(size_t)r * kC + c];
    float inv = 1.f / s;
    for (int c = 0; c < kC; ++c) m[(size_t)r * kC + c] *= inv;
    __syncthreads();
    s = 0.f;
    for (int k = 0; k < kC; ++k) s += m[(size_t)k * kC + r];
    inv = 1.f / s;
    for (int k = 0; k < kC; ++k) m[(size_t)k * kC + r] *= inv;
    __syncthreads();
  }
  float s = 0.f;                             // (m@L)[r,c] = sum_{k>=c} m[r,k]
  for (int c = kC - 1; c >= 0; --c) {
    s += m[(size_t)r * kC + c];
    mL[(size_t)r * kC + c] = (bf16)s;
  }
}

// ---- Pack B operands into WMMA fragment order ------------------------------
// lower[i,j] = P[j,i]. Modes: 0 = m^T, 1 = W_embed^T, 2 = [Wir;Whr|Wiz;Whz],
// 3 = [Win;Whn]. One wave packs one 32x16 tile.
__global__ void __launch_bounds__(256) pack_b_kernel(
    int mode, int Ktot, int Ntot,
    const float* __restrict__ S0, const float* __restrict__ S1,
    const float* __restrict__ S2, const float* __restrict__ S3,
    const float* __restrict__ P, bf16* __restrict__ out) {
  const int lane = threadIdx.x & 31;
  const int tile = blockIdx.x * 8 + (threadIdx.x >> 5);
  const int ktiles = Ktot >> 5, ntiles = Ntot >> 4;
  if (tile >= ktiles * ntiles) return;
  const int kt = tile / ntiles, nt = tile % ntiles;
  const int n  = nt * 16 + (lane & 15);
  const int kb = kt * 32 + ((lane & 16) ? 8 : 0);
  bf16* dst = out + (((size_t)kt * ntiles + nt) * 32 + lane) * 16;
  for (int e = 0; e < 16; ++e) {
    const int k = kb + e + ((e >= 8) ? 8 : 0);
    float v;
    if (mode == 0) {                       // B = m^T  (K=N=1024)
      v = S0[(size_t)n * kC + k];
    } else if (mode == 1) {                // B = W_embed^T (K=256, N=1024)
      v = S0[(size_t)n * kE + k];
    } else if (mode == 2) {                // stacked r|z gates, masked
      if (k < kC) {
        v = (n < kC) ? S0[(size_t)k * kC + n] * P[(size_t)n * kC + k]
                     : S1[(size_t)k * kC + (n - kC)] * P[(size_t)(n - kC) * kC + k];
      } else {
        const int kk = k - kC;
        v = (n < kC) ? S2[(size_t)kk * kC + n] * P[(size_t)n * kC + kk]
                     : S3[(size_t)kk * kC + (n - kC)] * P[(size_t)(n - kC) * kC + kk];
      }
    } else {                               // stacked n gate, masked
      if (k < kC) v = S0[(size_t)k * kC + n] * P[(size_t)n * kC + k];
      else {
        const int kk = k - kC;
        v = S1[(size_t)kk * kC + n] * P[(size_t)n * kC + kk];
      }
    }
    dst[e] = (bf16)v;
  }
}

// ---- P = mL @ m^T  (1024x1024x1024, f32 out), 2x-unrolled ping-pong --------
__global__ void __launch_bounds__(256) gemm_p_kernel(
    const bf16* __restrict__ A, const bf16* __restrict__ Bp,
    float* __restrict__ C) {
  const int lane = threadIdx.x & 31;
  const int wave = blockIdx.x * 8 + (threadIdx.x >> 5);
  const int mt = wave >> 4;        // 64 m-tiles
  const int ng = wave & 15;        // 16 groups of 64 cols
  const int ntiles = kC >> 4;
  constexpr int KT = kC / 32;      // 32 (even)
  v8f acc[4] = {};
  v16bf a0 = load_a_frag(A, kC, mt * 16, 0, lane);
  v16bf b0[4];
#pragma unroll
  for (int j = 0; j < 4; ++j) b0[j] = load_b_frag(Bp, ntiles, 0, ng * 4 + j, lane);
  for (int kt = 0; kt < KT; kt += 2) {
    v16bf a1 = load_a_frag(A, kC, mt * 16, (kt + 1) * 32, lane);
    v16bf b1[4];
#pragma unroll
    for (int j = 0; j < 4; ++j) b1[j] = load_b_frag(Bp, ntiles, kt + 1, ng * 4 + j, lane);
#pragma unroll
    for (int j = 0; j < 4; ++j) acc[j] = WMMA_BF16(a0, b0[j], acc[j]);
    if (kt + 2 < KT) {
      a0 = load_a_frag(A, kC, mt * 16, (kt + 2) * 32, lane);
#pragma unroll
      for (int j = 0; j < 4; ++j) b0[j] = load_b_frag(Bp, ntiles, kt + 2, ng * 4 + j, lane);
    }
#pragma unroll
    for (int j = 0; j < 4; ++j) acc[j] = WMMA_BF16(a1, b1[j], acc[j]);
  }
  const int row0 = mt * 16 + ((lane & 16) ? 8 : 0);
  const int col  = lane & 15;
#pragma unroll
  for (int j = 0; j < 4; ++j) {
    const int cb = (ng * 4 + j) * 16 + col;
#pragma unroll
    for (int g = 0; g < 8; ++g)
      C[(size_t)(row0 + g) * kC + cb] = acc[j][g];
  }
}

// ---- embedding gather: preembed (bf16) + rawembed . W_out[:E] --------------
__global__ void __launch_bounds__(256) embed_gather_kernel(
    const int* __restrict__ cidx, const int* __restrict__ labels,
    const float* __restrict__ embedM, const float* __restrict__ deltaM,
    const float* __restrict__ Wout,
    bf16* __restrict__ pre, float* __restrict__ re_dot) {
  const int lane = threadIdx.x & 31;
  const int i = blockIdx.x * 8 + (threadIdx.x >> 5);   // row = t*256 + b
  if (i >= kTB) return;
  const int t = i >> 8, b = i & 255;
  const int c = cidx[b * kT + t];
  const float fl = (float)labels[b * kT + t];
  const float fla = fabsf(fl);
  float dot = 0.f;
  const int e0 = lane * 8;
#pragma unroll
  for (int e = 0; e < 8; ++e) {
    const float em = embedM[(size_t)c * kE + e0 + e];
    const float de = deltaM[(size_t)c * kE + e0 + e];
    pre[(size_t)i * kE + e0 + e] = (bf16)(fla * em + fl * de);
    dot += em * Wout[e0 + e];
  }
  dot *= fla;
  for (int off = 16; off; off >>= 1) dot += __shfl_xor(dot, off, 32);
  if (lane == 0) re_dot[i] = dot;
}

// ---- X = preembed @ W_embed^T + b_embed  (bf16 out), 2x-unrolled -----------
__global__ void __launch_bounds__(256) gemm_embed_kernel(
    const bf16* __restrict__ pre, const bf16* __restrict__ Wep,
    const float* __restrict__ bemb, bf16* __restrict__ X) {
  const int lane = threadIdx.x & 31;
  const int wave = blockIdx.x * 8 + (threadIdx.x >> 5);
  const int mt = wave >> 4;        // 3200 m-tiles
  const int ng = wave & 15;        // N = 1024
  if (mt * 16 >= kTB) return;
  const int ntiles = kC >> 4;
  constexpr int KT = kE / 32;      // 8 (even)
  v8f acc[4] = {};
  v16bf a0 = load_a_frag(pre, kE, mt * 16, 0, lane);
  v16bf b0[4];
#pragma unroll
  for (int j = 0; j < 4; ++j) b0[j] = load_b_frag(Wep, ntiles, 0, ng * 4 + j, lane);
  for (int kt = 0; kt < KT; kt += 2) {
    v16bf a1 = load_a_frag(pre, kE, mt * 16, (kt + 1) * 32, lane);
    v16bf b1[4];
#pragma unroll
    for (int j = 0; j < 4; ++j) b1[j] = load_b_frag(Wep, ntiles, kt + 1, ng * 4 + j, lane);
#pragma unroll
    for (int j = 0; j < 4; ++j) acc[j] = WMMA_BF16(a0, b0[j], acc[j]);
    if (kt + 2 < KT) {
      a0 = load_a_frag(pre, kE, mt * 16, (kt + 2) * 32, lane);
#pragma unroll
      for (int j = 0; j < 4; ++j) b0[j] = load_b_frag(Wep, ntiles, kt + 2, ng * 4 + j, lane);
    }
#pragma unroll
    for (int j = 0; j < 4; ++j) acc[j] = WMMA_BF16(a1, b1[j], acc[j]);
  }
  const int row0 = mt * 16 + ((lane & 16) ? 8 : 0);
  const int col  = lane & 15;
#pragma unroll
  for (int j = 0; j < 4; ++j) {
    const int cb = (ng * 4 + j) * 16 + col;
#pragma unroll
    for (int g = 0; g < 8; ++g)
      X[(size_t)(row0 + g) * kC + cb] = (bf16)(acc[j][g] + bemb[cb]);
  }
}

__global__ void __launch_bounds__(256) zero_h_kernel(float* hf, bf16* hbf) {
  const int i = blockIdx.x * 256 + threadIdx.x;
  if (i < kB * kC) { hf[i] = 0.f; hbf[i] = (bf16)0.f; }
}

// ---- GRU step, phase A: [x_t ; h] @ [Wir;Whr | Wiz;Whz]  (K=2048, N=2048) --
// 8 waves/block share one n-group: B tiles async-staged to LDS, even/odd k
// bound statically to buffer 0/1 (copy-free ping-pong). Epilogue: r=sigmoid
// -> rh (bf16), z=sigmoid (f32). Block 0 gathers relevant[t] from h_{t-1}.
__global__ void __launch_bounds__(256) step_a_kernel(
    const bf16* __restrict__ Xall, int t,
    const bf16* __restrict__ hbf, const float* __restrict__ hf,
    const bf16* __restrict__ Brz,
    bf16* __restrict__ rh, float* __restrict__ zb,
    const int* __restrict__ cidx, float* __restrict__ relevant) {
  __shared__ __align__(16) bf16 lds_b[2][2048];     // 2 x 4KB double buffer
  const int tid  = threadIdx.x;
  const int lane = tid & 31;
  const int ngrp  = blockIdx.x & 31;                // shared by all 8 waves
  const int mhalf = blockIdx.x >> 5;
  const int mt = mhalf * 8 + (tid >> 5);
  const int ntiles = (2 * kC) >> 4;                 // 128
  const bf16* Xt = Xall + (size_t)t * kB * kC;

  if (blockIdx.x == 0) {                            // h_{t-1} gather (h not written here)
    relevant[t * kB + tid] = hf[(size_t)tid * kC + cidx[tid * kT + t]];
  }

  stage_async16(Brz + ((size_t)0 * ntiles + ngrp * 4) * 512 + tid * 8, &lds_b[0][tid * 8]);
  v8f acc[4] = {};
  v16bf a0 = load_a_frag(Xt, kC, mt * 16, 0, lane);
  v16bf a1;

  for (int kt = 0; kt < 64; kt += 2) {
    // even phase: consume buf0, stage kt+1 -> buf1
    wait_async0();
    __syncthreads();
    stage_async16(Brz + ((size_t)(kt + 1) * ntiles + ngrp * 4) * 512 + tid * 8,
                  &lds_b[1][tid * 8]);
    a1 = (kt + 1 < 32) ? load_a_frag(Xt,  kC, mt * 16, (kt + 1) * 32, lane)
                       : load_a_frag(hbf, kC, mt * 16, (kt + 1 - 32) * 32, lane);
    mma4_lds(lds_b[0], a0, lane, acc);
    // odd phase: consume buf1, stage kt+2 -> buf0
    wait_async0();
    __syncthreads();
    if (kt + 2 < 64) {
      stage_async16(Brz + ((size_t)(kt + 2) * ntiles + ngrp * 4) * 512 + tid * 8,
                    &lds_b[0][tid * 8]);
      a0 = (kt + 2 < 32) ? load_a_frag(Xt,  kC, mt * 16, (kt + 2) * 32, lane)
                         : load_a_frag(hbf, kC, mt * 16, (kt + 2 - 32) * 32, lane);
    }
    mma4_lds(lds_b[1], a1, lane, acc);
  }
  const int row0 = mt * 16 + ((lane & 16) ? 8 : 0);
  const int col  = lane & 15;
#pragma unroll
  for (int j = 0; j < 4; ++j) {
    const int cb = (ngrp * 4 + j) * 16 + col;
#pragma unroll
    for (int g = 0; g < 8; ++g) {
      const int row = row0 + g;
      const float v = sigmoidf_(acc[j][g]);
      if (cb < kC) {                                // r gate -> r*h
        const size_t idx = (size_t)row * kC + cb;
        rh[idx] = (bf16)(v * hf[idx]);
      } else {                                      // z gate
        zb[(size_t)row * kC + (cb - kC)] = v;
      }
    }
  }
}

// ---- GRU step, phase B: [x_t ; r*h] @ [Win;Whn]  (K=2048, N=1024) ----------
__global__ void __launch_bounds__(256) step_b_kernel(
    const bf16* __restrict__ Xall, int t,
    const bf16* __restrict__ rh, const bf16* __restrict__ Bn,
    const float* __restrict__ zb,
    float* __restrict__ hf, bf16* __restrict__ hbf) {
  __shared__ __align__(16) bf16 lds_b[2][2048];
  const int tid  = threadIdx.x;
  const int lane = tid & 31;
  const int ngrp  = blockIdx.x & 15;
  const int mhalf = blockIdx.x >> 4;
  const int mt = mhalf * 8 + (tid >> 5);
  const int ntiles = kC >> 4;                       // 64
  const bf16* Xt = Xall + (size_t)t * kB * kC;

  stage_async16(Bn + ((size_t)0 * ntiles + ngrp * 4) * 512 + tid * 8, &lds_b[0][tid * 8]);
  v8f acc[4] = {};
  v16bf a0 = load_a_frag(Xt, kC, mt * 16, 0, lane);
  v16bf a1;

  for (int kt = 0; kt < 64; kt += 2) {
    wait_async0();
    __syncthreads();
    stage_async16(Bn + ((size_t)(kt + 1) * ntiles + ngrp * 4) * 512 + tid * 8,
                  &lds_b[1][tid * 8]);
    a1 = (kt + 1 < 32) ? load_a_frag(Xt, kC, mt * 16, (kt + 1) * 32, lane)
                       : load_a_frag(rh, kC, mt * 16, (kt + 1 - 32) * 32, lane);
    mma4_lds(lds_b[0], a0, lane, acc);
    wait_async0();
    __syncthreads();
    if (kt + 2 < 64) {
      stage_async16(Bn + ((size_t)(kt + 2) * ntiles + ngrp * 4) * 512 + tid * 8,
                    &lds_b[0][tid * 8]);
      a0 = (kt + 2 < 32) ? load_a_frag(Xt, kC, mt * 16, (kt + 2) * 32, lane)
                         : load_a_frag(rh, kC, mt * 16, (kt + 2 - 32) * 32, lane);
    }
    mma4_lds(lds_b[1], a1, lane, acc);
  }
  const int row0 = mt * 16 + ((lane & 16) ? 8 : 0);
  const int col  = lane & 15;
#pragma unroll
  for (int j = 0; j < 4; ++j) {
    const int cb = (ngrp * 4 + j) * 16 + col;
#pragma unroll
    for (int g = 0; g < 8; ++g) {
      const size_t idx = (size_t)(row0 + g) * kC + cb;
      const float nv = tanhf(acc[j][g]);
      const float z  = zb[idx];
      const float hn = hf[idx] * z + (1.f - z) * nv;
      hf[idx]  = hn;
      hbf[idx] = (bf16)hn;
    }
  }
}

// ---- masked BCE-with-logits, deterministic two-stage reduction -------------
__global__ void __launch_bounds__(256) bce_partial_kernel(
    const float* __restrict__ re_dot, const float* __restrict__ relevant,
    const int* __restrict__ labels, const float* __restrict__ Wout,
    const float* __restrict__ bout, float* __restrict__ partials) {
  __shared__ float ssum[8];
  const int i = blockIdx.x * 256 + threadIdx.x;
  float contrib = 0.f;
  if (i < kTB) {
    const int t = i >> 8, b = i & 255;
    const int lb = labels[b * kT + t];
    const float y   = lb > 0 ? (float)lb : 0.f;
    const float msk = (lb != 0) ? 1.f : 0.f;
    const float lg  = re_dot[i] + relevant[i] * Wout[kE] + bout[0];
    const float raw = fmaxf(lg, 0.f) - lg * y + log1pf(expf(-fabsf(lg)));
    contrib = raw * msk;
  }
  for (int off = 16; off; off >>= 1) contrib += __shfl_xor(contrib, off, 32);
  if ((threadIdx.x & 31) == 0) ssum[threadIdx.x >> 5] = contrib;
  __syncthreads();
  if (threadIdx.x == 0) {
    float s = 0.f;
    for (int w = 0; w < 8; ++w) s += ssum[w];
    partials[blockIdx.x] = s;
  }
}

__global__ void __launch_bounds__(256) bce_final_kernel(
    const float* __restrict__ partials, float* __restrict__ out) {
  __shared__ float ssum[8];
  float v = (threadIdx.x < kTB / 256) ? partials[threadIdx.x] : 0.f;
  for (int off = 16; off; off >>= 1) v += __shfl_xor(v, off, 32);
  if ((threadIdx.x & 31) == 0) ssum[threadIdx.x >> 5] = v;
  __syncthreads();
  if (threadIdx.x == 0) {
    float s = 0.f;
    for (int w = 0; w < 8; ++w) s += ssum[w];
    out[0] = s / (float)kTB;
  }
}

// ---------------------------------------------------------------------------
extern "C" void kernel_launch(void* const* d_in, const int* in_sizes, int n_in,
                              void* d_out, int out_size, void* d_ws, size_t ws_size,
                              hipStream_t stream) {
  const int*   cidx   = (const int*)d_in[0];
  const int*   labels = (const int*)d_in[1];
  const int*   epoch  = (const int*)d_in[2];
  const float* perm   = (const float*)d_in[3];
  const float* embedM = (const float*)d_in[4];
  const float* deltaM = (const float*)d_in[5];
  const float* Wemb   = (const float*)d_in[6];
  const float* bemb   = (const float*)d_in[7];
  const float* Wir    = (const float*)d_in[8];
  const float* Whr    = (const float*)d_in[9];
  const float* Wiz    = (const float*)d_in[10];
  const float* Whz    = (const float*)d_in[11];
  const float* Win    = (const float*)d_in[12];
  const float* Whn    = (const float*)d_in[13];
  const float* Wout   = (const float*)d_in[14];
  const float* bout   = (const float*)d_in[15];
  float* out = (float*)d_out;

  // workspace carve (~155 MB, every buffer rewritten each call)
  char* w = (char*)d_ws;
  auto carve = [&](size_t bytes) { char* p = w; w += (bytes + 255) & ~(size_t)255; return p; };
  float* m      = (float*)carve((size_t)kC * kC * 4);
  float* P      = (float*)carve((size_t)kC * kC * 4);
  bf16*  mLbf   = (bf16*)carve((size_t)kC * kC * 2);
  bf16*  mTp    = (bf16*)carve((size_t)kC * kC * 2);
  bf16*  Brz    = (bf16*)carve((size_t)2 * kC * 2 * kC * 2);
  bf16*  Bn     = (bf16*)carve((size_t)2 * kC * kC * 2);
  bf16*  Wep    = (bf16*)carve((size_t)kE * kC * 2);
  bf16*  pre    = (bf16*)carve((size_t)kTB * kE * 2);
  bf16*  X      = (bf16*)carve((size_t)kTB * kC * 2);
  float* re_dot = (float*)carve((size_t)kTB * 4);
  float* relev  = (float*)carve((size_t)kTB * 4);
  float* hf     = (float*)carve((size_t)kB * kC * 4);
  bf16*  hbf    = (bf16*)carve((size_t)kB * kC * 2);
  bf16*  rh     = (bf16*)carve((size_t)kB * kC * 2);
  float* zb     = (float*)carve((size_t)kB * kC * 4);
  float* parts  = (float*)carve(256 * 4);

  // 1. Sinkhorn + reverse-cumsum (m@L) -> mL (bf16 A-operand)
  sinkhorn_kernel<<<1, 1024, 0, stream>>>(perm, epoch, m, mLbf);
  // 2. pack B = m^T, then P = mL @ m^T  (lower = P^T)
  pack_b_kernel<<<256, 256, 0, stream>>>(0, kC, kC, m, nullptr, nullptr, nullptr, nullptr, mTp);
  gemm_p_kernel<<<128, 256, 0, stream>>>(mLbf, mTp, P);
  // 3. pack masked GRU weights + W_embed^T
  pack_b_kernel<<<1024, 256, 0, stream>>>(2, 2 * kC, 2 * kC, Wir, Wiz, Whr, Whz, P, Brz);
  pack_b_kernel<<<512, 256, 0, stream>>>(3, 2 * kC, kC, Win, Whn, nullptr, nullptr, P, Bn);
  pack_b_kernel<<<64, 256, 0, stream>>>(1, kE, kC, Wemb, nullptr, nullptr, nullptr, nullptr, Wep);
  // 4. embedding gather + output-head partial dot
  embed_gather_kernel<<<6400, 256, 0, stream>>>(cidx, labels, embedM, deltaM, Wout, pre, re_dot);
  // 5. X = preembed @ W_embed^T + b_embed  (51200x1024, bf16)
  gemm_embed_kernel<<<6400, 256, 0, stream>>>(pre, Wep, bemb, X);
  // 6. GRU recurrence: 200 steps x 2 fused K=2048 WMMA GEMMs
  zero_h_kernel<<<(kB * kC) / 256, 256, 0, stream>>>(hf, hbf);
  for (int t = 0; t < kT; ++t) {
    step_a_kernel<<<64, 256, 0, stream>>>(X, t, hbf, hf, Brz, rh, zb, cidx, relev);
    step_b_kernel<<<32, 256, 0, stream>>>(X, t, rh, Bn, zb, hf, hbf);
  }
  // 7. masked BCE mean (deterministic reduction)
  bce_partial_kernel<<<kTB / 256, 256, 0, stream>>>(re_dot, relev, labels, Wout, bout, parts);
  bce_final_kernel<<<1, 256, 0, stream>>>(parts, out);
}
